// AdaptiveTemporalContrastEnhancement_41051297415246
// MI455X (gfx1250) — matrix operand order repaired and checked by reading, not compiled, not run on hardware
//
#include <hip/hip_runtime.h>
#include <hip/hip_bf16.h>

// Adaptive temporal-contrast attention for MI455X (gfx1250).
// delta_c is constant along the softmax axis => softmax-invariant => omitted.
//
// All matmuls run on v_wmma_f32_16x16x32_bf16 (bf16 operands, f32 accum).
//   1) cvt:   X, Wq/Wk/Wv/Wo -> bf16 copies in d_ws
//   2) qkv:   register-blocked 64x32 tiles/wave; Qb,Kb row-major, Vt transposed
//   3) attn:  8 waves/block share one (bt,h); K/V tiles double-buffered into
//             LDS via global_load_async_to_lds_b128 + s_wait_asynccnt;
//             flash-style online softmax; P transposed through LDS
//   4) oproj: register-blocked Out = Z Wo^T + bo (f32)

#define DIM      512
#define HEADS    8
#define HEAD_DIM 64
#define NTOK     1024
#define BT       16   // B*T

typedef __attribute__((ext_vector_type(16))) __bf16 v16bf;
typedef __attribute__((ext_vector_type(8)))  __bf16 v8bf;
typedef __attribute__((ext_vector_type(8)))  float  v8f;
typedef __attribute__((ext_vector_type(4)))  float  v4f;

static __device__ inline v16bf cat8(v8bf lo, v8bf hi) {
  return __builtin_shufflevector(lo, hi, 0,1,2,3,4,5,6,7,8,9,10,11,12,13,14,15);
}

static __device__ inline v8f wmma_bf16(v16bf a, v16bf b, v8f c) {
  return __builtin_amdgcn_wmma_f32_16x16x32_bf16(false, a, false, b, (short)0, c,
                                                 false, false);
}

// A-fragment (16x32 bf16): lane row = l&15; chunks at k = 16*half and 16*half+8.
static __device__ inline v16bf load_afrag(const __bf16* row, int half) {
  v8bf lo = *(const v8bf*)(row + half * 16);
  v8bf hi = *(const v8bf*)(row + half * 16 + 8);
  return cat8(lo, hi);
}

// gfx1250 async DMA: global -> LDS, 16B per lane, tracked by ASYNCcnt.
// LDS offset = low 32 bits of the generic pointer (LDS aperture keeps the
// workgroup-relative offset in addr[31:0]).
static __device__ inline void async_load_b128(void* lds_ptr, const void* gptr) {
  unsigned lds_off = (unsigned)(size_t)lds_ptr;
  asm volatile("global_load_async_to_lds_b128 %0, %1, off"
               :: "v"(lds_off), "v"(gptr) : "memory");
}
static __device__ inline void wait_async_le2() {
  asm volatile("s_wait_asynccnt 0x2" ::: "memory");
}
static __device__ inline void wait_async_0() {
  asm volatile("s_wait_asynccnt 0x0" ::: "memory");
}

// ---------------------------------------------------------------------------
// Kernel 1: fp32 -> bf16 conversion of X and the four weight matrices.
// ---------------------------------------------------------------------------
__global__ void cvt_kernel(const float* __restrict__ X,
                           const float* __restrict__ Wq, const float* __restrict__ Wk,
                           const float* __restrict__ Wv, const float* __restrict__ Wo,
                           __bf16* __restrict__ Xb,
                           __bf16* __restrict__ Wqb, __bf16* __restrict__ Wkb,
                           __bf16* __restrict__ Wvb, __bf16* __restrict__ Wob) {
  const long long NX = (long long)BT * NTOK * DIM;
  const long long NW = (long long)DIM * DIM;
  const long long total = NX + 4 * NW;
  long long stride = (long long)gridDim.x * blockDim.x;
  for (long long i = (long long)blockIdx.x * blockDim.x + threadIdx.x;
       i < total; i += stride) {
    if (i < NX) {
      Xb[i] = (__bf16)X[i];
    } else {
      long long j = i - NX;
      int w = (int)(j / NW);
      long long k = j % NW;
      const float* src = (w == 0) ? Wq : (w == 1) ? Wk : (w == 2) ? Wv : Wo;
      __bf16* dst      = (w == 0) ? Wqb : (w == 1) ? Wkb : (w == 2) ? Wvb : Wob;
      dst[k] = (__bf16)src[k];
    }
  }
}

// ---------------------------------------------------------------------------
// Kernel 2: QKV projections, register-blocked 4x2 (64x32 output per wave).
//  which=0/1 (Q/K): A = W rows (e), B = X rows (n): C rows=e, cols=n.
//  which=2   (V):   A = X rows (n), B = W rows (e): C rows=n, cols=e -> Vt.
// ---------------------------------------------------------------------------
__global__ void qkv_kernel(const __bf16* __restrict__ Xb,
                           const __bf16* __restrict__ Wqb, const __bf16* __restrict__ Wkb,
                           const __bf16* __restrict__ Wvb,
                           const float* __restrict__ bq, const float* __restrict__ bk,
                           const float* __restrict__ bv,
                           __bf16* __restrict__ Qb, __bf16* __restrict__ Kb,
                           __bf16* __restrict__ Vt) {
  const int wave = threadIdx.x >> 5;
  const int lane = threadIdx.x & 31;
  const int half = lane >> 4;
  const int l16  = lane & 15;
  const int bt   = blockIdx.y;
  const int which = blockIdx.z;                 // 0=Q 1=K 2=V
  const int tid   = blockIdx.x * 4 + wave;      // 0..255 wave-tiles/(bt,which)

  const __bf16* W    = (which == 0) ? Wqb : (which == 1) ? Wkb : Wvb;
  const float*  bias = (which == 0) ? bq  : (which == 1) ? bk  : bv;
  const __bf16* Xrow = Xb + (long long)bt * NTOK * DIM;

  const v8f zero8 = {0.f, 0.f, 0.f, 0.f, 0.f, 0.f, 0.f, 0.f};

  if (which < 2) {
    // 8 e-blocks (64) x 32 n-blocks (32)
    const int ebase = (tid & 7) * 64;
    const int nbase = (tid >> 3) * 32;
    const __bf16* Ar[4]; const __bf16* Br[2];
    #pragma unroll
    for (int i = 0; i < 4; ++i) Ar[i] = W + (long long)(ebase + i * 16 + l16) * DIM;
    #pragma unroll
    for (int j = 0; j < 2; ++j) Br[j] = Xrow + (long long)(nbase + j * 16 + l16) * DIM;
    v8f acc[4][2];
    #pragma unroll
    for (int i = 0; i < 4; ++i)
      #pragma unroll
      for (int j = 0; j < 2; ++j) acc[i][j] = zero8;
    #pragma unroll 2
    for (int k0 = 0; k0 < DIM; k0 += 32) {
      v16bf a[4], b[2];
      #pragma unroll
      for (int i = 0; i < 4; ++i) a[i] = load_afrag(Ar[i] + k0, half);
      #pragma unroll
      for (int j = 0; j < 2; ++j) b[j] = *(const v16bf*)(Br[j] + k0 + half * 16);
      #pragma unroll
      for (int i = 0; i < 4; ++i)
        #pragma unroll
        for (int j = 0; j < 2; ++j) acc[i][j] = wmma_bf16(a[i], b[j], acc[i][j]);
    }
    __bf16* Out = (which == 0) ? Qb : Kb;
    #pragma unroll
    for (int i = 0; i < 4; ++i) {
      v4f b0 = *(const v4f*)(bias + ebase + i * 16 + 8 * half);
      v4f b1 = *(const v4f*)(bias + ebase + i * 16 + 8 * half + 4);
      #pragma unroll
      for (int j = 0; j < 2; ++j) {
        v8bf o;
        #pragma unroll
        for (int r = 0; r < 8; ++r)
          o[r] = (__bf16)(acc[i][j][r] + ((r < 4) ? b0[r] : b1[r - 4]));
        __bf16* dst = Out + ((long long)bt * NTOK + nbase + j * 16 + l16) * DIM
                    + ebase + i * 16 + 8 * half;
        *(v8bf*)dst = o;
      }
    }
  } else {
    // 16 e-blocks (32) x 16 n-blocks (64); C rows = n, cols = e -> Vt[e][n]
    const int ebase = (tid & 15) * 32;
    const int nbase = (tid >> 4) * 64;
    const __bf16* Ar[4]; const __bf16* Br[2];
    #pragma unroll
    for (int i = 0; i < 4; ++i) Ar[i] = Xrow + (long long)(nbase + i * 16 + l16) * DIM;
    #pragma unroll
    for (int j = 0; j < 2; ++j) Br[j] = W + (long long)(ebase + j * 16 + l16) * DIM;
    v8f acc[4][2];
    #pragma unroll
    for (int i = 0; i < 4; ++i)
      #pragma unroll
      for (int j = 0; j < 2; ++j) acc[i][j] = zero8;
    #pragma unroll 2
    for (int k0 = 0; k0 < DIM; k0 += 32) {
      v16bf a[4], b[2];
      #pragma unroll
      for (int i = 0; i < 4; ++i) a[i] = load_afrag(Ar[i] + k0, half);
      #pragma unroll
      for (int j = 0; j < 2; ++j) b[j] = *(const v16bf*)(Br[j] + k0 + half * 16);
      #pragma unroll
      for (int i = 0; i < 4; ++i)
        #pragma unroll
        for (int j = 0; j < 2; ++j) acc[i][j] = wmma_bf16(a[i], b[j], acc[i][j]);
    }
    #pragma unroll
    for (int j = 0; j < 2; ++j) {
      float bsc = bias[ebase + j * 16 + l16];
      #pragma unroll
      for (int i = 0; i < 4; ++i) {
        v8bf o;
        #pragma unroll
        for (int r = 0; r < 8; ++r) o[r] = (__bf16)(acc[i][j][r] + bsc);
        __bf16* dst = Vt + ((long long)bt * DIM + ebase + j * 16 + l16) * NTOK
                    + nbase + i * 16 + 8 * half;
        *(v8bf*)dst = o;
      }
    }
  }
}

// ---------------------------------------------------------------------------
// Kernel 3: attention. 8 waves/block share one (bt,h) and cover 128 Q rows.
// K (32x64) and V (64x32) tiles are double-buffered into LDS with the gfx1250
// async DMA; per 32-col step: 4 WMMAs for S, online softmax (shfl_xor within
// 16-lane halves), P transposed through LDS, 4 WMMAs accumulating Z.
// ---------------------------------------------------------------------------
__global__ void attn_kernel(const __bf16* __restrict__ Qb,
                            const __bf16* __restrict__ Kb,
                            const __bf16* __restrict__ Vt,
                            __bf16* __restrict__ Zb) {
  __shared__ __bf16 kbuf[2][32][64];            // 8 KB  K tile (m-local, d)
  __shared__ __bf16 vbuf[2][64][32];            // 8 KB  V tile (d, m-local)
  __shared__ __bf16 pbuf[8][16][32];            // 8 KB  per-wave P staging
  const int tid  = threadIdx.x;
  const int wave = tid >> 5;
  const int lane = tid & 31;
  const int half = lane >> 4;
  const int l16  = lane & 15;
  const int blk  = blockIdx.x;                  // 16 bt * 8 h * 8 ngroups
  const int bt     = blk >> 6;
  const int h      = (blk >> 3) & 7;
  const int nbase  = (blk & 7) * 128 + wave * 16;

  const __bf16* Kbase = Kb + (long long)bt * NTOK * DIM + h * HEAD_DIM;
  const __bf16* Vbase = Vt + ((long long)bt * DIM + h * HEAD_DIM) * NTOK;

  // Per-thread async staging assignment: K = 32 rows x 8 chunks of 16B,
  // V = 64 rows x 4 chunks of 16B (256 chunks each; 1 per thread each).
  const int krow = tid >> 3, kcol = (tid & 7) * 8;
  const int vrow = tid >> 2, vcol = (tid & 3) * 8;
  auto stage = [&](int buf, int m0) {
    async_load_b128(&kbuf[buf][krow][kcol],
                    Kbase + (long long)(m0 + krow) * DIM + kcol);
    async_load_b128(&vbuf[buf][vrow][vcol],
                    Vbase + (long long)vrow * NTOK + m0 + vcol);
  };

  // Preload the two Q A-fragments (K-dim = Dh = 64 -> two k-steps of 32).
  const __bf16* Qrow = Qb + ((long long)bt * NTOK + nbase + l16) * DIM + h * HEAD_DIM;
  v16bf qa0 = load_afrag(Qrow, half);
  v16bf qa1 = load_afrag(Qrow + 32, half);

  const v8f zero8 = {0.f, 0.f, 0.f, 0.f, 0.f, 0.f, 0.f, 0.f};
  v8f zacc[4] = {zero8, zero8, zero8, zero8};
  float rmax[8], rsum[8];
  #pragma unroll
  for (int r = 0; r < 8; ++r) { rmax[r] = -1e30f; rsum[r] = 0.f; }
  const float scale = 0.125f;                   // 1/sqrt(64)

  stage(0, 0);                                  // prologue: tile 0

  for (int m0 = 0; m0 < NTOK; m0 += 32) {
    const int cur = (m0 >> 5) & 1;
    if (m0 + 32 < NTOK) {                       // prefetch next tile
      stage(cur ^ 1, m0 + 32);
      wait_async_le2();                         // current tile resident
    } else {
      wait_async_0();
    }
    __syncthreads();

    // --- S tiles: rows n, cols m (two 16-col subtiles) ---
    v8f s[2];
    #pragma unroll
    for (int sub = 0; sub < 2; ++sub) {
      v8f a = zero8;
      a = wmma_bf16(qa0, *(const v16bf*)&kbuf[cur][sub * 16 + l16][half * 16], a);
      a = wmma_bf16(qa1, *(const v16bf*)&kbuf[cur][sub * 16 + l16][32 + half * 16], a);
      #pragma unroll
      for (int r = 0; r < 8; ++r) a[r] *= scale;
      s[sub] = a;
    }
    // --- online softmax per row r (row = r + 8*half, replicated per half) ---
    float fac[8];
    #pragma unroll
    for (int r = 0; r < 8; ++r) {
      float m = fmaxf(s[0][r], s[1][r]);
      #pragma unroll
      for (int off = 1; off < 16; off <<= 1) m = fmaxf(m, __shfl_xor(m, off, 32));
      float nm = fmaxf(rmax[r], m);
      fac[r] = __expf(rmax[r] - nm);
      rmax[r] = nm;
      float p0 = __expf(s[0][r] - nm);
      float p1 = __expf(s[1][r] - nm);
      s[0][r] = p0; s[1][r] = p1;
      float ps = p0 + p1;
      #pragma unroll
      for (int off = 1; off < 16; off <<= 1) ps += __shfl_xor(ps, off, 32);
      rsum[r] = rsum[r] * fac[r] + ps;
    }
    #pragma unroll
    for (int t = 0; t < 4; ++t)
      #pragma unroll
      for (int r = 0; r < 8; ++r) zacc[t][r] *= fac[r];

    // --- transpose P through LDS: C layout -> A-fragment layout ---
    #pragma unroll
    for (int sub = 0; sub < 2; ++sub)
      #pragma unroll
      for (int r = 0; r < 8; ++r)
        pbuf[wave][r + 8 * half][sub * 16 + l16] = (__bf16)s[sub][r];
    v8bf plo = *(const v8bf*)&pbuf[wave][l16][half * 16];
    v8bf phi = *(const v8bf*)&pbuf[wave][l16][half * 16 + 8];
    v16bf pa = cat8(plo, phi);

    // --- Z += P V from the LDS V tile ---
    #pragma unroll
    for (int t = 0; t < 4; ++t)
      zacc[t] = wmma_bf16(pa, *(const v16bf*)&vbuf[cur][t * 16 + l16][half * 16],
                          zacc[t]);
    __syncthreads();                            // all waves done with buf `cur`
  }

  // --- normalize and store Z (bf16, row-major [bt,n, h*64+d]) ---
  __bf16* Zrow = Zb + ((long long)bt * NTOK + nbase) * DIM + h * HEAD_DIM;
  #pragma unroll
  for (int r = 0; r < 8; ++r) {
    float inv = 1.0f / rsum[r];
    #pragma unroll
    for (int t = 0; t < 4; ++t)
      Zrow[(long long)(r + 8 * half) * DIM + t * 16 + l16] = (__bf16)(zacc[t][r] * inv);
  }
}

// ---------------------------------------------------------------------------
// Kernel 4: output projection, register-blocked 4x2 (64x32 per wave).
// A = Wo rows (e), B = Z rows (n): C rows=e, cols=n -> packed v4f stores.
// ---------------------------------------------------------------------------
__global__ void oproj_kernel(const __bf16* __restrict__ Zb,
                             const __bf16* __restrict__ Wob,
                             const float* __restrict__ bo,
                             float* __restrict__ Out) {
  const int wave = threadIdx.x >> 5;
  const int lane = threadIdx.x & 31;
  const int half = lane >> 4;
  const int l16  = lane & 15;
  const int bt   = blockIdx.y;
  const int tid  = blockIdx.x * 4 + wave;
  const int ebase = (tid & 7) * 64;
  const int nbase = (tid >> 3) * 32;

  const __bf16* Ar[4]; const __bf16* Br[2];
  #pragma unroll
  for (int i = 0; i < 4; ++i) Ar[i] = Wob + (long long)(ebase + i * 16 + l16) * DIM;
  #pragma unroll
  for (int j = 0; j < 2; ++j)
    Br[j] = Zb + ((long long)bt * NTOK + nbase + j * 16 + l16) * DIM;

  const v8f zero8 = {0.f, 0.f, 0.f, 0.f, 0.f, 0.f, 0.f, 0.f};
  v8f acc[4][2];
  #pragma unroll
  for (int i = 0; i < 4; ++i)
    #pragma unroll
    for (int j = 0; j < 2; ++j) acc[i][j] = zero8;
  #pragma unroll 2
  for (int k0 = 0; k0 < DIM; k0 += 32) {
    v16bf a[4], b[2];
    #pragma unroll
    for (int i = 0; i < 4; ++i) a[i] = load_afrag(Ar[i] + k0, half);
    #pragma unroll
    for (int j = 0; j < 2; ++j) b[j] = *(const v16bf*)(Br[j] + k0 + half * 16);
    #pragma unroll
    for (int i = 0; i < 4; ++i)
      #pragma unroll
      for (int j = 0; j < 2; ++j) acc[i][j] = wmma_bf16(a[i], b[j], acc[i][j]);
  }
  #pragma unroll
  for (int i = 0; i < 4; ++i) {
    v4f b0 = *(const v4f*)(bo + ebase + i * 16 + 8 * half);
    v4f b1 = *(const v4f*)(bo + ebase + i * 16 + 8 * half + 4);
    #pragma unroll
    for (int j = 0; j < 2; ++j) {
      float* dst = Out + ((long long)bt * NTOK + nbase + j * 16 + l16) * DIM
                 + ebase + i * 16 + 8 * half;
      v4f o0, o1;
      #pragma unroll
      for (int r = 0; r < 4; ++r) {
        o0[r] = acc[i][j][r] + b0[r];
        o1[r] = acc[i][j][r + 4] + b1[r];
      }
      *(v4f*)dst = o0;
      *(v4f*)(dst + 4) = o1;
    }
  }
}

// ---------------------------------------------------------------------------
extern "C" void kernel_launch(void* const* d_in, const int* in_sizes, int n_in,
                              void* d_out, int out_size, void* d_ws, size_t ws_size,
                              hipStream_t stream) {
  (void)in_sizes; (void)n_in; (void)out_size; (void)ws_size;
  const float* X  = (const float*)d_in[0];
  const float* Wq = (const float*)d_in[1];
  const float* bq = (const float*)d_in[2];
  const float* Wk = (const float*)d_in[3];
  const float* bk = (const float*)d_in[4];
  const float* Wv = (const float*)d_in[5];
  const float* bv = (const float*)d_in[6];
  const float* Wo = (const float*)d_in[7];
  const float* bo = (const float*)d_in[8];
  float* Out = (float*)d_out;

  char* ws = (char*)d_ws;
  const size_t NXB = (size_t)BT * NTOK * DIM * 2;   // 16 MiB per activation buf
  const size_t NWB = (size_t)DIM * DIM * 2;         // 512 KiB per weight
  __bf16* Xb  = (__bf16*)(ws);
  __bf16* Wqb = (__bf16*)(ws + NXB);
  __bf16* Wkb = (__bf16*)(ws + NXB + NWB);
  __bf16* Wvb = (__bf16*)(ws + NXB + 2 * NWB);
  __bf16* Wob = (__bf16*)(ws + NXB + 3 * NWB);
  __bf16* Qb  = (__bf16*)(ws + NXB + 4 * NWB);
  __bf16* Kb  = (__bf16*)(ws + 2 * NXB + 4 * NWB);
  __bf16* Vt  = (__bf16*)(ws + 3 * NXB + 4 * NWB);
  __bf16* Zb  = (__bf16*)(ws + 4 * NXB + 4 * NWB);  // total ~82 MiB of d_ws

  cvt_kernel<<<4096, 256, 0, stream>>>(X, Wq, Wk, Wv, Wo, Xb, Wqb, Wkb, Wvb, Wob);

  dim3 gqkv(64, BT, 3);           // 256 wave-tiles/(bt,mat), 4 waves/block
  qkv_kernel<<<gqkv, 128, 0, stream>>>(Xb, Wqb, Wkb, Wvb, bq, bk, bv, Qb, Kb, Vt);

  attn_kernel<<<1024, 256, 0, stream>>>(Qb, Kb, Vt, Zb);   // (bt,h,ngroup) blocks

  dim3 gout(64, BT);
  oproj_kernel<<<gout, 128, 0, stream>>>(Zb, Wob, bo, Out);
}
